// AttentionBlock_53506702574052
// MI455X (gfx1250) — matrix operand order repaired and checked by reading, not compiled, not run on hardware
//
#include <hip/hip_runtime.h>

// ---------------- problem constants ----------------
#define BATCH 2
#define SEQ   2048
#define HID   2048
#define NHEAD 16
#define HDIM  128
#define SCALE 0.088388347648318447f   // 1/sqrt(128)

typedef __bf16 bf16_t;
typedef __attribute__((ext_vector_type(16))) __bf16 v16bf;
typedef __attribute__((ext_vector_type(8)))  __bf16 v8bf;
typedef __attribute__((ext_vector_type(8)))  float  v8f;

static __device__ const v8f kZero8 = {0.f,0.f,0.f,0.f,0.f,0.f,0.f,0.f};

// ---------------- async global->LDS path (gfx1250 ASYNCcnt) ----------------
#if __has_builtin(__builtin_amdgcn_global_load_async_to_lds_b128) && \
    __has_builtin(__builtin_amdgcn_s_wait_asynccnt)
#define HAVE_ASYNC_LDS 1
#else
#define HAVE_ASYNC_LDS 0
#endif

#if HAVE_ASYNC_LDS
// builtin signature (from compiler diagnostic): pointers to 128-bit int vectors
typedef int v4i_t __attribute__((vector_size(16)));
__device__ inline void async_cp16(void* lds, const void* g) {
  // GLOBAL_LOAD_ASYNC_TO_LDS_B128: 16B per lane, DMA into LDS, tracked by ASYNCcnt
  __builtin_amdgcn_global_load_async_to_lds_b128((v4i_t*)g, (v4i_t*)lds, 0, 0);
}
__device__ inline void async_wait0() { __builtin_amdgcn_s_wait_asynccnt(0); }
#endif

// ---------------- WMMA helpers ----------------
__device__ inline v8f wmma_bf16(v16bf a, v16bf b, v8f c) {
  // D = A(16x32 bf16) x B(32x16 bf16) + C(16x16 f32)
  return __builtin_amdgcn_wmma_f32_16x16x32_bf16(
      /*neg_a=*/false, a, /*neg_b=*/false, b,
      /*c_mod=*/(short)0, c, /*reuse_a=*/false, /*reuse_b=*/false);
}

// A fragment (16x32): row-major source, leading dim ld (elements).
__device__ inline v16bf load_frag_a(const bf16_t* base, int ld, int row0, int k0) {
  int lane = threadIdx.x & 31;
  int row  = row0 + (lane & 15);
  int k    = k0 + ((lane >> 4) << 3);
  const bf16_t* p = base + (size_t)row * (size_t)ld + k;
  union { v16bf v; v8bf h[2]; } u;
  u.h[0] = *(const v8bf*)(p);
  u.h[1] = *(const v8bf*)(p + 16);
  return u.v;
}

// B fragment (32x16): column n of B is a contiguous row of `base` (ld elements).
__device__ inline v16bf load_frag_b(const bf16_t* base, int ld, int col0, int k0) {
  int lane = threadIdx.x & 31;
  int col  = col0 + (lane & 15);
  int k    = k0 + ((lane >> 4) << 3);
  const bf16_t* p = base + (size_t)col * (size_t)ld + k;
  union { v16bf v; v8bf h[2]; } u;
  u.h[0] = *(const v8bf*)(p);
  u.h[1] = *(const v8bf*)(p + 16);
  return u.v;
}

// ---------------- f32 -> bf16 convert ----------------
__global__ void cvt_f32_bf16(const float* __restrict__ in, bf16_t* __restrict__ out, long n) {
  long i = (long)blockIdx.x * blockDim.x + threadIdx.x;
  long stride = (long)gridDim.x * blockDim.x;
  for (; i < n; i += stride) out[i] = (bf16_t)in[i];
}

// LDS strides (padded to break bank conflicts on 16B fragment reads)
#define LDA 40    // tiles with 32-element K rows
#define LDK 136   // K tile rows of 128 elements

// ---------------- Kernel 1: QKV GEMM + RoPE + scatter ----------------
__global__ void __launch_bounds__(256)
qkv_gemm_rope(const bf16_t* __restrict__ xb, const bf16_t* __restrict__ wq,
              const float* __restrict__ fc,
              bf16_t* __restrict__ Qb, bf16_t* __restrict__ Kb, bf16_t* __restrict__ Vt) {
  __shared__ bf16_t As[2][64 * LDA];
  __shared__ bf16_t Bs[2][128 * LDA];

  int tid  = threadIdx.x;
  int w    = tid >> 5, lane = tid & 31;
  int wm   = w & 3, wn = w >> 2;
  int mB   = blockIdx.x * 64;
  int nB   = blockIdx.y * 128;

  // cooperative-copy geometry: A tile 64x32 (1 chunk/thread), B tile 128x32 (2 chunks)
  int ar  = tid >> 2,          ac  = (tid & 3) << 3;
  int br0 = tid >> 2,          bc0 = (tid & 3) << 3;
  int br1 = (tid + 256) >> 2,  bc1 = ((tid + 256) & 3) << 3;
  const bf16_t* Ag  = xb + (size_t)(mB + ar)  * HID + ac;
  const bf16_t* Bg0 = wq + (size_t)(nB + br0) * HID + bc0;
  const bf16_t* Bg1 = wq + (size_t)(nB + br1) * HID + bc1;

  v8f acc[4];
#pragma unroll
  for (int i = 0; i < 4; ++i) acc[i] = kZero8;

  // prologue: stage k-chunk 0
#if HAVE_ASYNC_LDS
  async_cp16(&As[0][ar  * LDA + ac],  Ag);
  async_cp16(&Bs[0][br0 * LDA + bc0], Bg0);
  async_cp16(&Bs[0][br1 * LDA + bc1], Bg1);
  async_wait0();
#else
  *(v8bf*)(&As[0][ar  * LDA + ac])  = *(const v8bf*)(Ag);
  *(v8bf*)(&Bs[0][br0 * LDA + bc0]) = *(const v8bf*)(Bg0);
  *(v8bf*)(&Bs[0][br1 * LDA + bc1]) = *(const v8bf*)(Bg1);
#endif
  __syncthreads();

  int buf = 0;
  for (int k0 = 0; k0 < HID; k0 += 32) {
    bool nxt = (k0 + 32) < HID;
#if HAVE_ASYNC_LDS
    if (nxt) {
      async_cp16(&As[buf ^ 1][ar  * LDA + ac],  Ag  + k0 + 32);
      async_cp16(&Bs[buf ^ 1][br0 * LDA + bc0], Bg0 + k0 + 32);
      async_cp16(&Bs[buf ^ 1][br1 * LDA + bc1], Bg1 + k0 + 32);
      __builtin_prefetch(Bg0 + k0 + 64, 0, 1);
      __builtin_prefetch(Bg1 + k0 + 64, 0, 1);
    }
#else
    v8bf an, bn0, bn1;
    if (nxt) {
      an  = *(const v8bf*)(Ag  + k0 + 32);
      bn0 = *(const v8bf*)(Bg0 + k0 + 32);
      bn1 = *(const v8bf*)(Bg1 + k0 + 32);
      __builtin_prefetch(Bg0 + k0 + 64, 0, 1);
      __builtin_prefetch(Bg1 + k0 + 64, 0, 1);
    }
#endif
    v16bf a = load_frag_a(&As[buf][0], LDA, wm * 16, 0);
#pragma unroll
    for (int nt = 0; nt < 4; ++nt) {
      v16bf b = load_frag_b(&Bs[buf][0], LDA, wn * 64 + nt * 16, 0);
      acc[nt] = wmma_bf16(a, b, acc[nt]);
    }
#if HAVE_ASYNC_LDS
    async_wait0();
#else
    if (nxt) {
      *(v8bf*)(&As[buf ^ 1][ar  * LDA + ac])  = an;
      *(v8bf*)(&Bs[buf ^ 1][br0 * LDA + bc0]) = bn0;
      *(v8bf*)(&Bs[buf ^ 1][br1 * LDA + bc1]) = bn1;
    }
#endif
    __syncthreads();
    buf ^= 1;
  }

  // epilogue: RoPE + scatter
  int n  = lane & 15;
  int mg = (lane >> 4) << 3;
#pragma unroll
  for (int nt = 0; nt < 4; ++nt) {
#pragma unroll
    for (int v = 0; v < 8; ++v) {
      float f     = acc[nt][v];
      float other = __shfl_xor(f, 1, 32);         // RoPE partner column (d ^ 1)
      int row   = mB + wm * 16 + mg + v;          // flattened (b, s)
      int col   = nB + wn * 64 + nt * 16 + n;     // in [0, 3H)
      int b     = row >> 11;
      int s     = row & (SEQ - 1);
      int which = col >> 11;                      // 0=q, 1=k, 2=v
      int hcol  = col & (HID - 1);
      int head  = hcol >> 7;
      int d     = hcol & (HDIM - 1);
      if (which == 2) {
        Vt[((size_t)(b * NHEAD + head) * HDIM + d) * SEQ + s] = (bf16_t)f;
      } else {
        const float* fcp = fc + ((size_t)s * (HDIM / 2) + (d >> 1)) * 4;
        float cv = fcp[0];                        // fc[s,p,0,0] = cos
        float sv = fcp[2];                        // fc[s,p,1,0] = sin
        float res = (d & 1) ? (other * sv + f * cv)
                            : (f * cv - other * sv);
        bf16_t* dst = which ? Kb : Qb;
        dst[((size_t)(b * NHEAD + head) * SEQ + s) * HDIM + d] = (bf16_t)res;
      }
    }
  }
}

// ---------------- Kernel 2: causal flash attention ----------------
__global__ void __launch_bounds__(256)
flash_attn(const bf16_t* __restrict__ Qb, const bf16_t* __restrict__ Kb,
           const bf16_t* __restrict__ Vt, bf16_t* __restrict__ Ob) {
  __shared__ bf16_t Ks[2][32 * LDK];    // 32 keys x 128 head-dim
  __shared__ bf16_t Vs[2][128 * LDA];   // 128 head-dim x 32 keys (V^T rows)
  __shared__ bf16_t Ps[8][16 * LDA];    // per-wave P tile 16x32

  int qt   = blockIdx.x;
  int bh   = blockIdx.y;
  int b    = bh >> 4;
  int h    = bh & (NHEAD - 1);
  int tid  = threadIdx.x;
  int w    = tid >> 5, lane = tid & 31;
  int n    = lane & 15;
  int mg   = (lane >> 4) << 3;
  int q0   = qt * 128 + w * 16;

  const bf16_t* qbase = Qb + (size_t)bh * SEQ * HDIM;
  const bf16_t* kbase = Kb + (size_t)bh * SEQ * HDIM;
  const bf16_t* vbase = Vt + (size_t)bh * HDIM * SEQ;
  bf16_t* pl = &Ps[w][0];

  // cooperative-copy geometry: K tile 32x128 (2 chunks), V tile 128x32 (2 chunks)
  int kr0 = tid >> 4,         kc0 = (tid & 15) << 3;
  int kr1 = (tid + 256) >> 4, kc1 = ((tid + 256) & 15) << 3;
  int vr0 = tid >> 2,         vc0 = (tid & 3) << 3;
  int vr1 = (tid + 256) >> 2, vc1 = ((tid + 256) & 3) << 3;

  v16bf qf[4];
#pragma unroll
  for (int c = 0; c < 4; ++c) qf[c] = load_frag_a(qbase, HDIM, q0, c * 32);

  v8f oacc[8];
#pragma unroll
  for (int j = 0; j < 8; ++j) oacc[j] = kZero8;
  float mrow[8], lrow[8];
#pragma unroll
  for (int v = 0; v < 8; ++v) { mrow[v] = -3.0e38f; lrow[v] = 0.f; }

  int nkv = (qt + 1) * 128;              // causal bound for this q tile

  // prologue: stage KV tile 0
#if HAVE_ASYNC_LDS
  async_cp16(&Ks[0][kr0 * LDK + kc0], kbase + (size_t)kr0 * HDIM + kc0);
  async_cp16(&Ks[0][kr1 * LDK + kc1], kbase + (size_t)kr1 * HDIM + kc1);
  async_cp16(&Vs[0][vr0 * LDA + vc0], vbase + (size_t)vr0 * SEQ + vc0);
  async_cp16(&Vs[0][vr1 * LDA + vc1], vbase + (size_t)vr1 * SEQ + vc1);
  async_wait0();
#else
  *(v8bf*)(&Ks[0][kr0 * LDK + kc0]) = *(const v8bf*)(kbase + (size_t)kr0 * HDIM + kc0);
  *(v8bf*)(&Ks[0][kr1 * LDK + kc1]) = *(const v8bf*)(kbase + (size_t)kr1 * HDIM + kc1);
  *(v8bf*)(&Vs[0][vr0 * LDA + vc0]) = *(const v8bf*)(vbase + (size_t)vr0 * SEQ + vc0);
  *(v8bf*)(&Vs[0][vr1 * LDA + vc1]) = *(const v8bf*)(vbase + (size_t)vr1 * SEQ + vc1);
#endif
  __syncthreads();

  int buf = 0;
  for (int kv0 = 0; kv0 < nkv; kv0 += 32) {
    bool nxt = (kv0 + 32) < nkv;
#if HAVE_ASYNC_LDS
    if (nxt) {
      const bf16_t* kg = kbase + (size_t)(kv0 + 32) * HDIM;
      async_cp16(&Ks[buf ^ 1][kr0 * LDK + kc0], kg + (size_t)kr0 * HDIM + kc0);
      async_cp16(&Ks[buf ^ 1][kr1 * LDK + kc1], kg + (size_t)kr1 * HDIM + kc1);
      async_cp16(&Vs[buf ^ 1][vr0 * LDA + vc0], vbase + (size_t)vr0 * SEQ + kv0 + 32 + vc0);
      async_cp16(&Vs[buf ^ 1][vr1 * LDA + vc1], vbase + (size_t)vr1 * SEQ + kv0 + 32 + vc1);
    }
#else
    v8bf ka, kb2, va, vb2;
    if (nxt) {
      const bf16_t* kg = kbase + (size_t)(kv0 + 32) * HDIM;
      ka  = *(const v8bf*)(kg + (size_t)kr0 * HDIM + kc0);
      kb2 = *(const v8bf*)(kg + (size_t)kr1 * HDIM + kc1);
      va  = *(const v8bf*)(vbase + (size_t)vr0 * SEQ + kv0 + 32 + vc0);
      vb2 = *(const v8bf*)(vbase + (size_t)vr1 * SEQ + kv0 + 32 + vc1);
    }
#endif

    // ---- S = Q K^T on the staged tile ----
    v8f s0 = kZero8, s1 = kZero8;
#pragma unroll
    for (int c = 0; c < 4; ++c) {
      v16bf b0 = load_frag_b(&Ks[buf][0], LDK, 0,  c * 32);
      v16bf b1 = load_frag_b(&Ks[buf][0], LDK, 16, c * 32);
      s0 = wmma_bf16(qf[c], b0, s0);
      s1 = wmma_bf16(qf[c], b1, s1);
    }

    // ---- streaming softmax (per-row stats via intra-16-lane shuffles) ----
#pragma unroll
    for (int v = 0; v < 8; ++v) {
      int qrow = q0 + mg + v;
      float a0 = s0[v] * SCALE;
      float a1 = s1[v] * SCALE;
      if (kv0 + n      > qrow) a0 = -1.0e9f;
      if (kv0 + 16 + n > qrow) a1 = -1.0e9f;
      float mx = fmaxf(a0, a1);
      mx = fmaxf(mx, __shfl_xor(mx, 1, 32));
      mx = fmaxf(mx, __shfl_xor(mx, 2, 32));
      mx = fmaxf(mx, __shfl_xor(mx, 4, 32));
      mx = fmaxf(mx, __shfl_xor(mx, 8, 32));
      float mnew  = fmaxf(mrow[v], mx);
      float alpha = __expf(mrow[v] - mnew);
      float p0 = __expf(a0 - mnew);
      float p1 = __expf(a1 - mnew);
      float rs = p0 + p1;
      rs += __shfl_xor(rs, 1, 32);
      rs += __shfl_xor(rs, 2, 32);
      rs += __shfl_xor(rs, 4, 32);
      rs += __shfl_xor(rs, 8, 32);
      lrow[v] = lrow[v] * alpha + rs;
      mrow[v] = mnew;
#pragma unroll
      for (int j = 0; j < 8; ++j) oacc[j][v] = oacc[j][v] * alpha;
      pl[(mg + v) * LDA + n]      = (bf16_t)p0;
      pl[(mg + v) * LDA + 16 + n] = (bf16_t)p1;
    }

    // ---- O += P V ----
    v16bf pf = load_frag_a(pl, LDA, 0, 0);
#pragma unroll
    for (int j = 0; j < 8; ++j) {
      v16bf vb = load_frag_b(&Vs[buf][0], LDA, j * 16, 0);
      oacc[j] = wmma_bf16(pf, vb, oacc[j]);
    }

#if HAVE_ASYNC_LDS
    async_wait0();
#else
    if (nxt) {
      *(v8bf*)(&Ks[buf ^ 1][kr0 * LDK + kc0]) = ka;
      *(v8bf*)(&Ks[buf ^ 1][kr1 * LDK + kc1]) = kb2;
      *(v8bf*)(&Vs[buf ^ 1][vr0 * LDA + vc0]) = va;
      *(v8bf*)(&Vs[buf ^ 1][vr1 * LDA + vc1]) = vb2;
    }
#endif
    __syncthreads();
    buf ^= 1;
  }

#pragma unroll
  for (int j = 0; j < 8; ++j) {
#pragma unroll
    for (int v = 0; v < 8; ++v) {
      int qrow = q0 + mg + v;
      float o = oacc[j][v] / lrow[v];
      Ob[((size_t)b * SEQ + qrow) * HID + h * HDIM + j * 16 + n] = (bf16_t)o;
    }
  }
}

// ---------------- Kernel 3: output projection ----------------
__global__ void __launch_bounds__(256)
out_gemm(const bf16_t* __restrict__ Ob, const bf16_t* __restrict__ wo,
         float* __restrict__ out) {
  __shared__ bf16_t As[2][64 * LDA];
  __shared__ bf16_t Bs[2][128 * LDA];

  int tid  = threadIdx.x;
  int w    = tid >> 5, lane = tid & 31;
  int wm   = w & 3, wn = w >> 2;
  int mB   = blockIdx.x * 64;
  int nB   = blockIdx.y * 128;

  int ar  = tid >> 2,          ac  = (tid & 3) << 3;
  int br0 = tid >> 2,          bc0 = (tid & 3) << 3;
  int br1 = (tid + 256) >> 2,  bc1 = ((tid + 256) & 3) << 3;
  const bf16_t* Ag  = Ob + (size_t)(mB + ar)  * HID + ac;
  const bf16_t* Bg0 = wo + (size_t)(nB + br0) * HID + bc0;
  const bf16_t* Bg1 = wo + (size_t)(nB + br1) * HID + bc1;

  v8f acc[4];
#pragma unroll
  for (int i = 0; i < 4; ++i) acc[i] = kZero8;

#if HAVE_ASYNC_LDS
  async_cp16(&As[0][ar  * LDA + ac],  Ag);
  async_cp16(&Bs[0][br0 * LDA + bc0], Bg0);
  async_cp16(&Bs[0][br1 * LDA + bc1], Bg1);
  async_wait0();
#else
  *(v8bf*)(&As[0][ar  * LDA + ac])  = *(const v8bf*)(Ag);
  *(v8bf*)(&Bs[0][br0 * LDA + bc0]) = *(const v8bf*)(Bg0);
  *(v8bf*)(&Bs[0][br1 * LDA + bc1]) = *(const v8bf*)(Bg1);
#endif
  __syncthreads();

  int buf = 0;
  for (int k0 = 0; k0 < HID; k0 += 32) {
    bool nxt = (k0 + 32) < HID;
#if HAVE_ASYNC_LDS
    if (nxt) {
      async_cp16(&As[buf ^ 1][ar  * LDA + ac],  Ag  + k0 + 32);
      async_cp16(&Bs[buf ^ 1][br0 * LDA + bc0], Bg0 + k0 + 32);
      async_cp16(&Bs[buf ^ 1][br1 * LDA + bc1], Bg1 + k0 + 32);
      __builtin_prefetch(Bg0 + k0 + 64, 0, 1);
      __builtin_prefetch(Bg1 + k0 + 64, 0, 1);
    }
#else
    v8bf an, bn0, bn1;
    if (nxt) {
      an  = *(const v8bf*)(Ag  + k0 + 32);
      bn0 = *(const v8bf*)(Bg0 + k0 + 32);
      bn1 = *(const v8bf*)(Bg1 + k0 + 32);
      __builtin_prefetch(Bg0 + k0 + 64, 0, 1);
      __builtin_prefetch(Bg1 + k0 + 64, 0, 1);
    }
#endif
    v16bf a = load_frag_a(&As[buf][0], LDA, wm * 16, 0);
#pragma unroll
    for (int nt = 0; nt < 4; ++nt) {
      v16bf b = load_frag_b(&Bs[buf][0], LDA, wn * 64 + nt * 16, 0);
      acc[nt] = wmma_bf16(a, b, acc[nt]);
    }
#if HAVE_ASYNC_LDS
    async_wait0();
#else
    if (nxt) {
      *(v8bf*)(&As[buf ^ 1][ar  * LDA + ac])  = an;
      *(v8bf*)(&Bs[buf ^ 1][br0 * LDA + bc0]) = bn0;
      *(v8bf*)(&Bs[buf ^ 1][br1 * LDA + bc1]) = bn1;
    }
#endif
    __syncthreads();
    buf ^= 1;
  }

  int n  = lane & 15;
  int mg = (lane >> 4) << 3;
#pragma unroll
  for (int nt = 0; nt < 4; ++nt) {
#pragma unroll
    for (int v = 0; v < 8; ++v) {
      int row = mB + wm * 16 + mg + v;
      int col = nB + wn * 64 + nt * 16 + n;
      out[(size_t)row * HID + col] = acc[nt][v];
    }
  }
}

// ---------------- host launcher ----------------
extern "C" void kernel_launch(void* const* d_in, const int* in_sizes, int n_in,
                              void* d_out, int out_size, void* d_ws, size_t ws_size,
                              hipStream_t stream) {
  (void)in_sizes; (void)n_in; (void)out_size; (void)ws_size;
  const float* x    = (const float*)d_in[0];
  // d_in[1] = attention_mask: causal mask is computed arithmetically (same values)
  const float* fc   = (const float*)d_in[2];
  const float* Wqkv = (const float*)d_in[3];
  const float* Wout = (const float*)d_in[4];

  const size_t nX  = (size_t)BATCH * SEQ * HID;   // 8.39M
  const size_t nWq = (size_t)3 * HID * HID;       // 12.6M
  const size_t nWo = (size_t)HID * HID;           // 4.19M

  bf16_t* xb  = (bf16_t*)d_ws;
  bf16_t* wqb = xb  + nX;
  bf16_t* wob = wqb + nWq;
  bf16_t* Qb  = wob + nWo;
  bf16_t* Kb  = Qb  + nX;
  bf16_t* Vt  = Kb  + nX;
  bf16_t* Ob  = Vt  + nX;          // total ~117 MB of workspace

  cvt_f32_bf16<<<2048, 256, 0, stream>>>(x,    xb,  (long)nX);
  cvt_f32_bf16<<<2048, 256, 0, stream>>>(Wqkv, wqb, (long)nWq);
  cvt_f32_bf16<<<2048, 256, 0, stream>>>(Wout, wob, (long)nWo);

  // QKV GEMM: M = B*S = 4096 (64 tiles), N = 3H = 6144 (48 tiles)
  qkv_gemm_rope<<<dim3(64, 48), 256, 0, stream>>>(xb, wqb, fc, Qb, Kb, Vt);

  // Flash attention: (S/128) q-tiles x (B*NH) head slots
  flash_attn<<<dim3(SEQ / 128, BATCH * NHEAD), 256, 0, stream>>>(Qb, Kb, Vt, Ob);

  // Output GEMM: M = 4096 (64 tiles), N = H = 2048 (16 tiles)
  out_gemm<<<dim3(64, 16), 256, 0, stream>>>(Ob, wob, (float*)d_out);
}
